// SpatialGcnRecon_42734924595612
// MI455X (gfx1250) — compile-verified, matrix operand order checked by reading, not run on hardware
//
#include <hip/hip_runtime.h>
#include <hip/hip_bf16.h>

// ---------------------------------------------------------------------------
// Fused SpatialGcnRecon for MI455X (gfx1250, wave32, WMMA bf16 16x16x32).
//
//   y[n, k*256+c, t, v] = sum_ci x[n,ci,t,v] * W[ci, k*256+c] + b[k*256+c]
//   out[n,c,t,w]        = sum_k sum_v y[n,k,c,t,v] * M[n,k,v,w]
//   M[n,k] = A[k] (k<3) ; lamda*B[n,k-3] (k>=3)
//
// One workgroup = (n, 16 out-channels, 16 timesteps).
// GEMM tile: M=96 (6k x 16c), N=512 (16t x 32 vpad), K=256 (8 steps of 32),
// bf16 inputs / f32 accumulate via v_wmma_f32_16x16x32_bf16.
// Staging stores are DWORD-packed (2 bf16 per ds_store_b32); conversion uses
// native __bf16 casts so clang emits hardware cvt instead of manual RNE.
// ---------------------------------------------------------------------------

typedef __attribute__((ext_vector_type(8)))  float        v8f;
typedef __attribute__((ext_vector_type(16))) __bf16       v16bf;
typedef __attribute__((ext_vector_type(4)))  unsigned int u32x4;

#define CW      1536   // OUT_C * K_NUM
#define VDIM    25
#define VP      32     // padded V
#define MROWS   96     // 6 * 16
#define APITCH  40     // bf16 elems per A row (32 + 8 pad) -> 20 u32
#define XPITCH  40     // bf16 elems per X column (32 + 8 pad) -> 20 u32
#define YPITCH  404    // f32 elems per Y row (400 + 4 pad)
#define MS_CNT  3750   // 6 * 25 * 25

// smem layout (bytes):
//   [0,     15000)  Ms   : 6x25x25 f32 graph matrices (B half pre-scaled by lamda)
//   [15000, 15384)  bS   : 96 f32 biases for this c-tile
//   [15392, ...  )  union:
//        stage:  As (96x40 bf16 = 7680 B) | Xs (512x40 bf16 = 40960 B)
//        agg  :  Ys (96x404 f32 = 155136 B)
#define SMEM_BYTES (15392 + 96 * YPITCH * 4)   // 170528

// Hardware bf16 convert + DWORD pack (2 bf16 per ds_store_b32).
__device__ __forceinline__ unsigned int pack2bf(float lo, float hi) {
  union { __bf16 b[2]; unsigned int u; } p;
  p.b[0] = (__bf16)lo;
  p.b[1] = (__bf16)hi;
  return p.u;
}

// Build a 16x32 bf16 WMMA fragment for this lane from LDS.
// Per ISA layout: lanes 0-15 need K in {0..7, 16..23}, lanes 16-31 need
// K in {8..15, 24..31} -> two ds_load_b128 per fragment (kb = (lane>>4)*8).
__device__ __forceinline__ v16bf ldfrag(const unsigned short* p) {
  union { v16bf v; u32x4 q[2]; } u;
  u.q[0] = *reinterpret_cast<const u32x4*>(p);
  u.q[1] = *reinterpret_cast<const u32x4*>(p + 16);
  return u.v;
}

__global__ __launch_bounds__(512, 1)
void gcn_fused(const float* __restrict__ x,    const float* __restrict__ Amat,
               const float* __restrict__ Bmat, const float* __restrict__ lamp,
               const float* __restrict__ Wg,   const float* __restrict__ bg,
               float* __restrict__ out)
{
  extern __shared__ char smem[];
  float*          Ms = reinterpret_cast<float*>(smem);
  float*          bS = reinterpret_cast<float*>(smem + 15000);
  char*           un = smem + 15392;
  unsigned short* As = reinterpret_cast<unsigned short*>(un);
  unsigned short* Xs = As + MROWS * APITCH;
  unsigned int*   As32 = reinterpret_cast<unsigned int*>(As);
  unsigned int*   Xs32 = reinterpret_cast<unsigned int*>(Xs);
  float*          Ys = reinterpret_cast<float*>(un);

  const int tid  = (int)threadIdx.x;
  const int lane = tid & 31;           // wave32
  const int wid  = tid >> 5;           // 0..15
  const int tb = (int)blockIdx.x;      // t-tile   0..3
  const int cb = (int)blockIdx.y;      // c-tile   0..15
  const int n  = (int)blockIdx.z;      // batch    0..31
  const int t0 = tb * 16;
  const int c0 = cb * 16;
  const float lamda = lamp[0];

  // ---- preload graph matrices (B pre-scaled by lamda) and biases ----------
  for (int e = tid; e < MS_CNT; e += 512) {
    int k = e / 625, r = e - k * 625;
    Ms[e] = (k < 3) ? Amat[k * 625 + r]
                    : lamda * Bmat[(n * 3 + (k - 3)) * 625 + r];
  }
  if (tid < MROWS) {
    int k = tid >> 4, cl = tid & 15;
    bS[tid] = bg[k * 256 + c0 + cl];
  }

  // ---- stage 1: GEMM  Y[96 x 512] = W^T[96 x 256] @ X[256 x 512] ----------
  const int wm   = wid >> 3;           // 0..1  -> 3 M-tiles each
  const int wn   = wid & 7;            // 0..7  -> 4 N-tiles each
  const int kb   = (lane >> 4) * 8;    // fragment K sub-base per half-wave
  const int lrow = lane & 15;

  const v8f vzero = {0.f, 0.f, 0.f, 0.f, 0.f, 0.f, 0.f, 0.f};
  v8f acc[3][4];
#pragma unroll
  for (int a = 0; a < 3; ++a)
#pragma unroll
    for (int bi = 0; bi < 4; ++bi) acc[a][bi] = vzero;

  // cooperative X-load mapping: one (ci-pair, t_local, v-half) per thread
  const int cip = tid >> 5;            // 0..15 -> channels {2cip, 2cip+1}
  const int tl  = (tid >> 1) & 15;     // 0..15
  const int vh  = tid & 1;             // v in [16*vh, 16*vh+16)

  for (int ks = 0; ks < 8; ++ks) {
    // --- A block: W[ci, k*256+c] -> bf16 pairs, rows = (k,c), cols = ci ---
#pragma unroll
    for (int i = 0; i < 3; ++i) {
      int e   = tid + i * 512;                 // 0..1535 exactly
      int row = e >> 4, kp = e & 15;
      int col = (row >> 4) * 256 + c0 + (row & 15);
      int kk  = ks * 32 + kp * 2;
      As32[row * 20 + kp] = pack2bf(Wg[(size_t)kk * CW + col],
                                    Wg[(size_t)(kk + 1) * CW + col]);
    }
    // --- X block: x[n, ci, t, v] -> bf16 pairs, column-major [(t,v)][ci] --
    const float* xA =
        x + (((size_t)n * 256 + (size_t)(ks * 32 + 2 * cip)) * 64 + (t0 + tl)) * VDIM;
    const float* xB = xA + (size_t)64 * VDIM;
    if (ks < 7) __builtin_prefetch(xA + (size_t)32 * 64 * VDIM, 0, 1);
#pragma unroll
    for (int vv = 0; vv < 16; ++vv) {
      int v = vh * 16 + vv;
      unsigned int pk = 0u;
      if (v < VDIM) pk = pack2bf(xA[v], xB[v]);
      Xs32[(tl * VP + v) * 20 + cip] = pk;
    }
    __syncthreads();

    // --- fragments + WMMA -------------------------------------------------
    v16bf af[3], bfr[4];
#pragma unroll
    for (int a = 0; a < 3; ++a) {
      int row = (wm * 3 + a) * 16 + lrow;
      af[a] = ldfrag(As + row * APITCH + kb);
    }
#pragma unroll
    for (int bi = 0; bi < 4; ++bi) {
      int col = (wn * 4 + bi) * 16 + lrow;
      bfr[bi] = ldfrag(Xs + col * XPITCH + kb);
    }
#pragma unroll
    for (int a = 0; a < 3; ++a)
#pragma unroll
      for (int bi = 0; bi < 4; ++bi)
        acc[a][bi] = __builtin_amdgcn_wmma_f32_16x16x32_bf16(
            false, af[a], false, bfr[bi], (short)0, acc[a][bi], false, false);
    __syncthreads();   // protect As/Xs before next iteration / Ys overwrite
  }

  // ---- dump accumulators (+bias) to LDS Y-tile, dropping padded v --------
  // C/D layout: VGPR j -> row 16*mt + j + 8*(lane>=16), col = lane&15.
  const int rofs = (lane >> 4) * 8;
#pragma unroll
  for (int a = 0; a < 3; ++a) {
    int mt = wm * 3 + a;
#pragma unroll
    for (int bi = 0; bi < 4; ++bi) {
      int nt   = wn * 4 + bi;
      int colp = nt * 16 + lrow;
      int tl2  = colp >> 5, v2 = colp & 31;
      if (v2 < VDIM) {
#pragma unroll
        for (int j = 0; j < 8; ++j) {
          int row = mt * 16 + rofs + j;
          Ys[row * YPITCH + tl2 * VDIM + v2] = acc[a][bi][j] + bS[row];
        }
      }
    }
  }
  __syncthreads();

  // ---- stage 2: graph aggregation out[c,t,w] = sum_{k,v} Y[(k,c)][t,v]*M --
  const int pairIdx = tid >> 1;          // 0..255 -> (c, t)
  const int half    = tid & 1;           // w-split: [0,13) / [13,25)
  const int c       = pairIdx >> 4;
  const int t       = pairIdx & 15;
  const int w0      = half ? 13 : 0;
  const int wcnt    = half ? 12 : 13;

  float sum[13];
#pragma unroll
  for (int w = 0; w < 13; ++w) sum[w] = 0.f;

  for (int k = 0; k < 6; ++k) {
    const float* yrow = Ys + (k * 16 + c) * YPITCH + t * VDIM;
    float yv[VDIM];
#pragma unroll
    for (int v = 0; v < VDIM; ++v) yv[v] = yrow[v];
    const float* mrow = Ms + k * 625 + w0;
#pragma unroll
    for (int v = 0; v < VDIM; ++v) {
#pragma unroll
      for (int w = 0; w < 13; ++w)
        if (w < wcnt) sum[w] = fmaf(yv[v], mrow[v * VDIM + w], sum[w]);
    }
  }

  float* op = out + (((size_t)n * 256 + (c0 + c)) * 64 + (t0 + t)) * VDIM + w0;
#pragma unroll
  for (int w = 0; w < 13; ++w)
    if (w < wcnt) op[w] = sum[w];
}

// second tuple element: pass A through
__global__ void copyA_kernel(const float* __restrict__ Amat, float* __restrict__ dst) {
  int i = (int)(blockIdx.x * 256 + threadIdx.x);
  if (i < 3 * 25 * 25) dst[i] = Amat[i];
}

extern "C" void kernel_launch(void* const* d_in, const int* in_sizes, int n_in,
                              void* d_out, int out_size, void* d_ws, size_t ws_size,
                              hipStream_t stream) {
  (void)in_sizes; (void)n_in; (void)out_size; (void)d_ws; (void)ws_size;
  const float* x    = (const float*)d_in[0];
  const float* Amat = (const float*)d_in[1];
  const float* Bmat = (const float*)d_in[2];
  const float* lam  = (const float*)d_in[3];
  const float* Wg   = (const float*)d_in[4];
  const float* bg   = (const float*)d_in[5];
  float* out = (float*)d_out;

  dim3 grid(4 /*t-tiles*/, 16 /*c-tiles*/, 32 /*n*/);
  gcn_fused<<<grid, dim3(512), (size_t)SMEM_BYTES, stream>>>(
      x, Amat, Bmat, lam, Wg, bg, out);

  const size_t xsum_elems = (size_t)32 * 256 * 64 * 25;   // 13,107,200
  copyA_kernel<<<(3 * 25 * 25 + 255) / 256, 256, 0, stream>>>(Amat, out + xsum_elems);
}
